// MASC_86930138071343
// MI455X (gfx1250) — matrix operand order repaired and checked by reading, not compiled, not run on hardware
//
#include <hip/hip_runtime.h>

// ---------------------------------------------------------------------------
// Per-sample masked 3x3 conv  ==  batched implicit GEMM (M=256, N=4096, K=2304)
// bf16 WMMA (v_wmma_f32_16x16x32_bf16) with f32 accumulation.
// Input patches staged via GLOBAL_LOAD_ASYNC_TO_LDS_B128 (ASYNCcnt), double
// buffered and pipelined one ci-chunk ahead. Chunk loop unrolled x2 so both
// LDS buffers are compile-time bases (immediate ds offsets, no cselect).
//
// Workspace layout (needs ~130 MiB):
//   [0, 2304)                 : softmax masks, f32 [B][9]
//   [4096, 4096+1179648)      : weights bf16 [9][Cout=256][Cin=256]
//   [1183744, +134217728)     : input bf16 NHWC [B][H][W][C]
// ---------------------------------------------------------------------------

typedef __attribute__((ext_vector_type(16))) __bf16        v16bf;
typedef __attribute__((ext_vector_type(8)))  float         v8f;
typedef __attribute__((ext_vector_type(4)))  unsigned int  u32x4;

#define SA_K  40              // padded K stride (elements) for A tiles in LDS
#define SB_K  40              // padded K stride (elements) for input patch
#define ATAP  (128 * SA_K)    // per-tap A tile stride

union UA { u32x4 q[2]; v16bf v; };

static __device__ __forceinline__ float bf2f(unsigned int h) {
  union { unsigned int u; float f; } x; x.u = h << 16; return x.f;
}
static __device__ __forceinline__ unsigned int f2bf(float f) {
  union { float f; unsigned int u; } x; x.f = f;
  unsigned int r = x.u + 0x7FFFu + ((x.u >> 16) & 1u);   // round-nearest-even
  return r >> 16;
}

// ---------------------------------------------------------------- softmax ---
__global__ void masc_softmax(const float* __restrict__ a, float* __restrict__ m) {
  int b = threadIdx.x;
  if (b >= 64) return;
  float v[9], mx = -1e30f;
#pragma unroll
  for (int j = 0; j < 9; ++j) { v[j] = a[b * 9 + j]; mx = fmaxf(mx, v[j]); }
  float s = 0.f;
#pragma unroll
  for (int j = 0; j < 9; ++j) { v[j] = __expf(v[j] - mx); s += v[j]; }
  float inv = 1.f / s;
#pragma unroll
  for (int j = 0; j < 9; ++j) m[b * 9 + j] = v[j] * inv;
}

// --------------------------------------------- weights: f32 OIHW -> bf16 ---
__global__ __launch_bounds__(256) void masc_pack_w(const float* __restrict__ w,
                                                   unsigned short* __restrict__ Wb) {
  int idx = blockIdx.x * 256 + threadIdx.x;           // 9*256*256 total
  int tap = idx >> 16;
  int rem = idx & 65535;
  int co  = rem >> 8, ci = rem & 255;
  Wb[idx] = (unsigned short)f2bf(w[(co * 256 + ci) * 9 + tap]);
}

// ------------------------------------- input: f32 NCHW -> bf16 NHWC (LDS) ---
__global__ __launch_bounds__(256) void masc_pack_x(const float* __restrict__ inp,
                                                   unsigned short* __restrict__ inT) {
  __shared__ float tile[128 * 65];                    // 33.3 KB, conflict-free
  int b = blockIdx.x >> 6, h = blockIdx.x & 63;
  unsigned short* dst = inT + (b * 4096 + h * 64) * 256;
#pragma unroll 1
  for (int half = 0; half < 2; ++half) {
    __syncthreads();
    for (int i = threadIdx.x; i < 128 * 64; i += 256) {
      int c = i >> 6, w = i & 63;                      // coalesced along w
      tile[c * 65 + w] = inp[((b * 256 + half * 128 + c) * 64 + h) * 64 + w];
    }
    __syncthreads();
    for (int i = threadIdx.x; i < 64 * 128; i += 256) {
      int w = i >> 7, c = i & 127;                     // coalesced along c
      dst[w * 256 + half * 128 + c] = (unsigned short)f2bf(tile[c * 65 + w]);
    }
  }
}

// ------------------------------------------------------------ main GEMM ----
// Block: 256 thr (8 waves). Tile: M=128 co x N=128 pixels (8h x 16w).
// Wave grid 2(M) x 4(N); each wave: 64x32 = 4x2 WMMA tiles, 8 x v8f acc.
__global__ __launch_bounds__(256) void masc_main(
    const unsigned short* __restrict__ inT,    // [B][H][W][C] bf16
    const unsigned short* __restrict__ Wb,     // [9][Cout][Cin] bf16
    const float* __restrict__ masks,           // [B][9]
    float* __restrict__ out)                   // [B][Cout][H][W] f32
{
  __shared__ unsigned short sA[3 * 128 * SA_K];     // 30720 B (3 taps)
  __shared__ unsigned short sB0[180 * SB_K];        // 14400 B, buffer 0
  __shared__ unsigned short sB1[180 * SB_K];        // 14400 B, buffer 1
  __shared__ unsigned short sScr[32 * 8];           // 512 B async OOB sink

  const int tid   = threadIdx.x;
  const int lane  = tid & 31;
  const int wid   = tid >> 5;
  const int wm    = wid & 1;
  const int wn    = wid >> 1;
  const int l16   = lane & 15;
  const int lhalf = lane >> 4;

  const int s   = blockIdx.x;
  const int th0 = (s >> 2) * 8;
  const int tw0 = (s & 3) * 16;
  const int co0 = blockIdx.y * 128;
  const int b   = blockIdx.z;

  float mk[9];
#pragma unroll
  for (int t = 0; t < 9; ++t) mk[t] = masks[b * 9 + t];

  // Per-lane fragment offsets per ISA VGPR layouts.
  int aoff[4];
#pragma unroll
  for (int mt = 0; mt < 4; ++mt)
    aoff[mt] = (wm * 64 + mt * 16 + l16) * SA_K + lhalf * 8;
  int pbase[2];
#pragma unroll
  for (int nt = 0; nt < 2; ++nt) {
    int nl = wn * 32 + nt * 16 + l16;
    pbase[nt] = ((nl >> 4) * 18 + (nl & 15)) * SB_K + lhalf * 8;
  }

  v8f acc[8];
#pragma unroll
  for (int i = 0; i < 8; ++i) acc[i] = (v8f)0.0f;

  const u32x4* __restrict__ inTv = (const u32x4*)inT;
  const unsigned int* __restrict__ Wbv = (const unsigned int*)Wb;
  const unsigned int scrAddr = (unsigned int)(size_t)(sScr + lane * 8);

  // Async-stage one ci-chunk of the 10x18 halo patch into LDS buffer `dst`.
  // Exactly 3 global_load_async_to_lds_b128 per wave (uniform -> safe
  // s_wait_asynccnt 0x3 pipelining; completions are in-order). OOB lanes:
  // async to scratch sink + ds_store zeros to the real slot (disjoint).
  auto stage_async = [&](unsigned short* dst, int ci0) {
#pragma unroll
    for (int k = 0; k < 3; ++k) {
      int i   = tid + k * 256;              // 0..767 (720 real + 48 dup tail)
      int pos = i >> 2; if (pos > 179) pos = 179;
      int sub = i & 3;
      int r = pos / 18, c = pos - r * 18;
      int gh = th0 - 1 + r, gw = tw0 - 1 + c;
      bool inb = ((unsigned)gh < 64u) && ((unsigned)gw < 64u);
      int ghc = inb ? gh : 0, gwc = inb ? gw : 0;     // clamped (safe) addr
      int gi = (((b * 4096 + ghc * 64 + gwc) << 8) + ci0) >> 3;
      unsigned long long gaddr = (unsigned long long)(size_t)(inTv + gi + sub);
      unsigned int laddr = inb
          ? (unsigned int)(size_t)(dst + pos * SB_K + sub * 8)
          : scrAddr;
      asm volatile("global_load_async_to_lds_b128 %0, %1, off"
                   :: "v"(laddr), "v"(gaddr) : "memory");
      if (!inb) *(u32x4*)(dst + pos * SB_K + sub * 8) = (u32x4)0u;
    }
  };

  // Stage 3 mask-scaled weight taps for row kh of the 3x3 kernel.
  auto stage_A = [&](int kh, int ci0) {
#pragma unroll 1
    for (int i = tid; i < 6144; i += 256) {
      int j   = i >> 11;                   // tap-in-row 0..2
      int rem = i & 2047;
      int col = rem >> 4;                  // co_local 0..127
      int cp  = rem & 15;                  // ci pair
      int tap = kh * 3 + j;
      unsigned int w2 = Wbv[((tap * 256 + co0 + col) << 7) + (ci0 >> 1) + cp];
      float m = mk[tap];
      unsigned int lo = f2bf(bf2f(w2 & 0xffffu) * m);
      unsigned int hi = f2bf(bf2f(w2 >> 16) * m);
      *(unsigned int*)(sA + (j * 128 + col) * SA_K + cp * 2) = lo | (hi << 16);
    }
  };

  // Compute 3 taps x (4M x 2N) WMMA against patch buffer `cur`, kernel row kh.
  auto compute_row = [&](const unsigned short* cur, int kh) {
#pragma unroll
    for (int j = 0; j < 3; ++j) {
      const unsigned short* sAt = sA + j * ATAP;
      UA a[4];
#pragma unroll
      for (int mt = 0; mt < 4; ++mt) {
        a[mt].q[0] = *(const u32x4*)(sAt + aoff[mt]);
        a[mt].q[1] = *(const u32x4*)(sAt + aoff[mt] + 16);
      }
      const int tOff = (kh * 18 + j) * SB_K;          // (dh*18+dw) shift
      UA bfr[2];
#pragma unroll
      for (int nt = 0; nt < 2; ++nt) {
        const unsigned short* p = cur + pbase[nt] + tOff;
        bfr[nt].q[0] = *(const u32x4*)(p);
        bfr[nt].q[1] = *(const u32x4*)(p + 16);
      }
#pragma unroll
      for (int mt = 0; mt < 4; ++mt)
#pragma unroll
        for (int nt = 0; nt < 2; ++nt)
          acc[mt * 2 + nt] = __builtin_amdgcn_wmma_f32_16x16x32_bf16(
              false, a[mt].v, false, bfr[nt].v, (short)0,
              acc[mt * 2 + nt], false, false);
    }
  };

  // One full ci-chunk: barrier, pipeline next chunk's DMA, stage+compute the
  // three kernel rows. cur/nxt are compile-time constants at each call site.
  auto process_chunk = [&](int ic, unsigned short* cur, unsigned short* nxt) {
    const int ci0 = ic * 32;
    __syncthreads();                        // chunk ic-1 readers done -> nxt free
    if (ic < 7) stage_async(nxt, ci0 + 32); // pipeline next chunk (ASYNCcnt +3)

    stage_A(0, ci0);                        // overlap VALU with async DMA
    if (ic < 7) asm volatile("s_wait_asynccnt 0x3" ::: "memory");
    else        asm volatile("s_wait_asynccnt 0x0" ::: "memory");
    __syncthreads();                        // publish sB(cur) + sA(kh=0)
    compute_row(cur, 0);

#pragma unroll
    for (int kh = 1; kh < 3; ++kh) {        // fully unrolled: imm ds offsets
      __syncthreads();                      // sA readers done
      stage_A(kh, ci0);
      __syncthreads();
      compute_row(cur, kh);
    }
  };

  // ---- prologue: issue chunk 0's async transfers ----
  stage_async(sB0, 0);

#pragma unroll 1
  for (int icp = 0; icp < 4; ++icp) {       // x2 unroll: static buffer bases
    process_chunk(2 * icp,     sB0, sB1);
    process_chunk(2 * icp + 1, sB1, sB0);
  }

  // ---- epilogue: D layout -> NCHW f32 stores (64B contiguous per v) ----
  const int h_base = th0 + wn * 2;
  const int w_out  = tw0 + l16;
#pragma unroll
  for (int mt = 0; mt < 4; ++mt) {
#pragma unroll
    for (int nt = 0; nt < 2; ++nt) {
      int h = h_base + nt;
#pragma unroll
      for (int v = 0; v < 8; ++v) {
        int co = co0 + wm * 64 + mt * 16 + lhalf * 8 + v;
        out[((b * 256 + co) << 12) + h * 64 + w_out] = acc[mt * 2 + nt][v];
      }
    }
  }
}

// ---------------------------------------------------------------------------
extern "C" void kernel_launch(void* const* d_in, const int* in_sizes, int n_in,
                              void* d_out, int out_size, void* d_ws, size_t ws_size,
                              hipStream_t stream) {
  const float* inp = (const float*)d_in[0];        // [64,256,64,64]
  const float* act = (const float*)d_in[1];        // [64,9]
  const float* cw  = (const float*)d_in[2];        // [256,256,3,3]
  float* out = (float*)d_out;

  char* ws = (char*)d_ws;
  float*          masks = (float*)ws;
  unsigned short* Wb    = (unsigned short*)(ws + 4096);
  unsigned short* inT   = (unsigned short*)(ws + 4096 + 9 * 256 * 256 * 2);

  masc_softmax<<<1, 64, 0, stream>>>(act, masks);
  masc_pack_w<<<(9 * 256 * 256) / 256, 256, 0, stream>>>(cw, Wb);
  masc_pack_x<<<64 * 64, 256, 0, stream>>>(inp, inT);
  masc_main<<<dim3(32, 2, 64), 256, 0, stream>>>(inT, Wb, masks, out);
}
